// SpikingTransformerEncoderLayer_25254407701018
// MI455X (gfx1250) — matrix operand order repaired and checked by reading, not compiled, not run on hardware
//
#include <hip/hip_runtime.h>
#include <cstdint>
#include <cstddef>

// ---------- types ----------
typedef __bf16 bf16;
typedef __bf16 bf16x4 __attribute__((ext_vector_type(4)));
typedef __bf16 bf16x8 __attribute__((ext_vector_type(8)));
typedef __bf16 v16bf  __attribute__((ext_vector_type(16)));
typedef float  v8f    __attribute__((ext_vector_type(8)));
typedef float  floatx4 __attribute__((ext_vector_type(4)));
typedef unsigned int uint32x4 __attribute__((ext_vector_type(4)));
typedef int    int32x4 __attribute__((ext_vector_type(4)));
typedef int    int32x8 __attribute__((ext_vector_type(8)));

#define DEVINL __device__ __forceinline__

#if __has_builtin(__builtin_amdgcn_tensor_load_to_lds) && __has_builtin(__builtin_amdgcn_s_wait_tensorcnt)
#define USE_TDM 1
#else
#define USE_TDM 0
#endif

DEVINL bf16 f2bf(float f) { return (bf16)f; }   // native cvt, RNE

DEVINL v16bf cat8(bf16x8 lo, bf16x8 hi) {
    return __builtin_shufflevector(lo, hi, 0,1,2,3,4,5,6,7,8,9,10,11,12,13,14,15);
}
DEVINL v8f v8zero() { v8f z = {0.f,0.f,0.f,0.f,0.f,0.f,0.f,0.f}; return z; }

DEVINL v8f wmma_bf16(v16bf a, v16bf b, v8f c) {
    // D = A(16x32 bf16) x B(32x16 bf16) + C(16x16 f32)
    return __builtin_amdgcn_wmma_f32_16x16x32_bf16(false, a, false, b, (short)0, c, false, false);
}

// A-fragment (16x32 bf16): lane holds 16 contiguous K at offset (lane>=16)*16
DEVINL v16bf load_afrag(const bf16* base, int stride, int lane) {
    int hf = lane >> 4, l = lane & 15;
    const bf16* p = base + l * stride + hf * 16;
    return cat8(*(const bf16x8*)(p), *(const bf16x8*)(p + 8));
}
// B-fragment (32x16 bf16): lane holds 16 contiguous K at offset (lane>=16)*16
DEVINL v16bf load_bfrag(const bf16* base, int stride, int lane) {
    int hf = lane >> 4, l = lane & 15;
    const bf16* p = base + l * stride + hf * 16;
    return cat8(*(const bf16x8*)(p), *(const bf16x8*)(p + 8));
}

#if USE_TDM
// ---------- Tensor Data Mover: 2D bf16 tile (tile_rows x tile_cols) global -> LDS ----------
// D# per CDNA5 ISA. global_addr = tile start; data_size=1 (2 bytes);
// tile_dim0 = cols, tile_dim1 = rows, tensor_dim0_stride = row stride (elements).
DEVINL void tdm_load_2d_bf16(const bf16* gtile, unsigned lds_byte_off,
                             int tile_rows, int tile_cols,
                             int tensor_rows, int tensor_cols, int row_stride)
{
    unsigned long long ga = (unsigned long long)(uintptr_t)gtile;
    uint32x4 g0;
    g0[0] = 1u;                                           // count=1, user descriptor
    g0[1] = lds_byte_off;                                 // lds_addr
    g0[2] = (unsigned)(ga & 0xFFFFFFFFu);                 // global_addr[31:0]
    g0[3] = (unsigned)((ga >> 32) & 0x01FFFFFFu) | (2u << 30); // addr[56:32] | type=2 (image)
    int32x8 g1;
    g1[0] = (1 << 16);                                    // wg_mask=0, data_size=1 (2B)
    g1[1] = (tensor_cols & 0xFFFF) << 16;                 // tensor_dim0[15:0]
    g1[2] = ((tensor_cols >> 16) & 0xFFFF) | ((tensor_rows & 0xFFFF) << 16); // dim0 hi | dim1 lo
    g1[3] = ((tensor_rows >> 16) & 0xFFFF) | ((tile_cols & 0xFFFF) << 16);   // dim1 hi | tile_dim0
    g1[4] = (tile_rows & 0xFFFF);                         // tile_dim1 (tile_dim2=0)
    g1[5] = row_stride;                                   // tensor_dim0_stride[31:0]
    g1[6] = 0;                                            // stride hi | dim1_stride lo
    g1[7] = 0;
    int32x4 z4 = {0, 0, 0, 0};
#if __clang_major__ >= 23
    int32x8 z8 = {0, 0, 0, 0, 0, 0, 0, 0};
    __builtin_amdgcn_tensor_load_to_lds(g0, g1, z4, z4, z8, 0);
#else
    __builtin_amdgcn_tensor_load_to_lds(g0, g1, z4, z4, 0);
#endif
}
DEVINL unsigned lds_offset_of(const void* p) {
    // generic LDS address: low 32 bits are the wave-relative LDS byte offset
    return (unsigned)(uintptr_t)p;
}
#endif

// ---------- fp32 -> bf16 conversion ----------
__global__ __launch_bounds__(256)
void cvt_bf16_kernel(const floatx4* __restrict__ in, bf16x4* __restrict__ out, int n4) {
    int i = blockIdx.x * 256 + threadIdx.x;
    if (i < n4) {
        floatx4 v = in[i];
        bf16x4 o;
        #pragma unroll
        for (int k = 0; k < 4; ++k) o[k] = f2bf(v[k]);
        out[i] = o;
    }
}

// ---------- generic bf16 GEMM: C[M,N] = A[M,K] * Bw[N,K]^T + bias ----------
// WG: 256 thr = 8 waves (2 M x 4 N), wave tile 64x32, WG tile 128x128, K step 32.
template<bool OUT_BF16>
__global__ __launch_bounds__(256)
void gemm_bf16_kernel(const bf16* __restrict__ A, const bf16* __restrict__ Bw,
                      const float* __restrict__ bias, void* __restrict__ Cout,
                      int M, int N, int K)
{
    __shared__ __attribute__((aligned(16))) bf16 As[128 * 32];
    __shared__ __attribute__((aligned(16))) bf16 Bs[128 * 32];
    const int tid = threadIdx.x, wave = tid >> 5, lane = tid & 31;
    const int wm = wave & 1, wn = wave >> 1;
    const int m0 = blockIdx.y * 128, n0 = blockIdx.x * 128;

    v8f acc[4][2];
    #pragma unroll
    for (int i = 0; i < 4; ++i) { acc[i][0] = v8zero(); acc[i][1] = v8zero(); }

    const int r = tid >> 1, c = (tid & 1) * 16;
    for (int k0 = 0; k0 < K; k0 += 32) {
#if USE_TDM
        // one wave drives the Tensor Data Mover for both tiles; TDM ignores EXEC,
        // a uniform scalar branch keeps other waves from duplicating the DMA.
        if (wave == 0) {
            tdm_load_2d_bf16(A  + (size_t)m0 * K + k0, lds_offset_of(As),
                             128, 32, M - m0, K - k0, K);
            tdm_load_2d_bf16(Bw + (size_t)n0 * K + k0, lds_offset_of(Bs),
                             128, 32, N - n0, K - k0, K);
            __builtin_amdgcn_s_wait_tensorcnt(0);
        }
        __syncthreads();
#else
        const bf16* ga = A  + (size_t)(m0 + r) * K + k0 + c;
        const bf16* gb = Bw + (size_t)(n0 + r) * K + k0 + c;
        *(bf16x8*)(As + r * 32 + c)     = *(const bf16x8*)(ga);
        *(bf16x8*)(As + r * 32 + c + 8) = *(const bf16x8*)(ga + 8);
        *(bf16x8*)(Bs + r * 32 + c)     = *(const bf16x8*)(gb);
        *(bf16x8*)(Bs + r * 32 + c + 8) = *(const bf16x8*)(gb + 8);
        if (k0 + 32 < K) {   // gfx1250 global_prefetch_b8 for next K tile
            __builtin_prefetch(ga + 32, 0, 0);
            __builtin_prefetch(gb + 32, 0, 0);
        }
        __syncthreads();
#endif
        v16bf bfr[2];
        #pragma unroll
        for (int j = 0; j < 2; ++j)
            bfr[j] = load_bfrag(Bs + (wn * 32 + j * 16) * 32, 32, lane);
        #pragma unroll
        for (int i = 0; i < 4; ++i) {
            v16bf af = load_afrag(As + (wm * 64 + i * 16) * 32, 32, lane);
            #pragma unroll
            for (int j = 0; j < 2; ++j)
                acc[i][j] = wmma_bf16(af, bfr[j], acc[i][j]);
        }
        __syncthreads();
    }

    const int hf = lane >> 4, l = lane & 15;
    #pragma unroll
    for (int j = 0; j < 2; ++j) {
        const int ncol = n0 + wn * 32 + j * 16 + l;
        const float bj = bias ? bias[ncol] : 0.f;
        #pragma unroll
        for (int i = 0; i < 4; ++i) {
            #pragma unroll
            for (int rr = 0; rr < 8; ++rr) {
                const int mrow = m0 + wm * 64 + i * 16 + rr + hf * 8;
                float vv = acc[i][j][rr] + bj;
                if (OUT_BF16) ((bf16*)Cout)[(size_t)mrow * N + ncol] = f2bf(vv);
                else          ((float*)Cout)[(size_t)mrow * N + ncol] = vv;
            }
        }
    }
}

// ---------- fused per-head attention ----------
// grid: B*H*4 blocks; 128 thr (4 waves); each wave owns 16 q rows.
__global__ __launch_bounds__(128)
void attn_kernel(const bf16* __restrict__ qkv, bf16* __restrict__ attn_out)
{
    const int E = 1024, HD = 64, Bb = 8, L3 = 3072;
    __shared__ __attribute__((aligned(16))) bf16 Vt[64 * 256];     // V transposed [d][key]
    __shared__ __attribute__((aligned(16))) bf16 Ps[4 * 16 * 256]; // probs [wave][row][key]
    const int blk = blockIdx.x;
    const int rb = blk & 3, bh = blk >> 2, h = bh & 15, b = bh >> 4;
    const int tid = threadIdx.x, wave = tid >> 5, lane = tid & 31;
    const int hf = lane >> 4, l = lane & 15;
    const int q0 = rb * 64 + wave * 16;

    // stage V (transposed) into LDS: contiguous global reads, scattered LDS writes
    for (int rr = tid; rr < 256; rr += 128) {
        const bf16* srcv = qkv + (size_t)(rr * Bb + b) * L3 + 2 * E + h * HD;
        #pragma unroll
        for (int cc = 0; cc < 64; cc += 8) {
            bf16x8 t8 = *(const bf16x8*)(srcv + cc);
            #pragma unroll
            for (int q = 0; q < 8; ++q) Vt[(cc + q) * 256 + rr] = t8[q];
        }
    }

    // Q A-fragments (head dim 64 = two K steps of 32)
    const bf16* qrow = qkv + (size_t)((q0 + l) * Bb + b) * L3 + h * HD;
    v16bf aq[2];
    #pragma unroll
    for (int kk = 0; kk < 2; ++kk) {
        const bf16* p = qrow + kk * 32 + hf * 16;
        aq[kk] = cat8(*(const bf16x8*)(p), *(const bf16x8*)(p + 8));
    }

    // scores = (Q K^T) * 1/sqrt(64), held in registers (16 tiles of 16 cols)
    v8f sc[16];
    #pragma unroll
    for (int nt = 0; nt < 16; ++nt) {
        v8f cacc = v8zero();
        #pragma unroll
        for (int kk = 0; kk < 2; ++kk) {
            const bf16* krow = qkv + (size_t)((nt * 16 + l) * Bb + b) * L3 + E + h * HD + kk * 32 + hf * 16;
            v16bf bk = cat8(*(const bf16x8*)(krow), *(const bf16x8*)(krow + 8));
            cacc = wmma_bf16(aq[kk], bk, cacc);
        }
        sc[nt] = cacc * 0.125f;
    }

    // softmax per row; row M=rr lives in lanes 0-15, M=rr+8 in lanes 16-31.
    // xor masks 1,2,4,8 reduce within each half-wave only (wave32).
    #pragma unroll
    for (int rr = 0; rr < 8; ++rr) {
        float mx = -3.4e38f;
        #pragma unroll
        for (int nt = 0; nt < 16; ++nt) mx = fmaxf(mx, sc[nt][rr]);
        #pragma unroll
        for (int d = 1; d < 16; d <<= 1) mx = fmaxf(mx, __shfl_xor(mx, d, 32));
        float sum = 0.f;
        #pragma unroll
        for (int nt = 0; nt < 16; ++nt) { float e = __expf(sc[nt][rr] - mx); sc[nt][rr] = e; sum += e; }
        #pragma unroll
        for (int d = 1; d < 16; d <<= 1) sum += __shfl_xor(sum, d, 32);
        float inv = 1.f / sum;
        #pragma unroll
        for (int nt = 0; nt < 16; ++nt)
            Ps[wave * 4096 + (rr + hf * 8) * 256 + nt * 16 + l] = f2bf(sc[nt][rr] * inv);
    }
    __syncthreads();

    // out = P @ V, K = 256 keys (8 steps of 32), N = 64 (4 tiles)
    v8f o[4];
    #pragma unroll
    for (int j = 0; j < 4; ++j) o[j] = v8zero();
    #pragma unroll
    for (int kt = 0; kt < 8; ++kt) {
        const bf16* prow = Ps + wave * 4096 + l * 256 + kt * 32 + hf * 16;
        v16bf ap = cat8(*(const bf16x8*)(prow), *(const bf16x8*)(prow + 8));
        #pragma unroll
        for (int j = 0; j < 4; ++j) {
            const bf16* vp = Vt + (j * 16 + l) * 256 + kt * 32 + hf * 16;
            v16bf bv = cat8(*(const bf16x8*)(vp), *(const bf16x8*)(vp + 8));
            o[j] = wmma_bf16(ap, bv, o[j]);
        }
    }
    #pragma unroll
    for (int j = 0; j < 4; ++j) {
        #pragma unroll
        for (int rr = 0; rr < 8; ++rr) {
            int srow = q0 + rr + hf * 8;
            attn_out[(size_t)(srow * Bb + b) * E + h * HD + j * 16 + l] = f2bf(o[j][rr]);
        }
    }
}

// ---------- residual + LayerNorm ----------
__global__ __launch_bounds__(256)
void ln_residual_kernel(const float* __restrict__ a, const float* __restrict__ res,
                        const float* __restrict__ g, const float* __restrict__ be,
                        float* __restrict__ out)
{
    const int E = 1024;
    const int row = blockIdx.x, tid = threadIdx.x;
    __shared__ float redS[8], redQ[8];
    const float* pa = a + (size_t)row * E;
    const float* pr = res + (size_t)row * E;
    float vals[4]; float s = 0.f, q = 0.f;
    #pragma unroll
    for (int i = 0; i < 4; ++i) {
        int idx = tid + i * 256;
        float v = pa[idx] + pr[idx];
        vals[i] = v; s += v; q += v * v;
    }
    #pragma unroll
    for (int d = 1; d < 32; d <<= 1) { s += __shfl_xor(s, d, 32); q += __shfl_xor(q, d, 32); }
    int wave = tid >> 5, lane = tid & 31;
    if (lane == 0) { redS[wave] = s; redQ[wave] = q; }
    __syncthreads();
    s = 0.f; q = 0.f;
    #pragma unroll
    for (int w = 0; w < 8; ++w) { s += redS[w]; q += redQ[w]; }
    float mean = s * (1.f / E);
    float var  = q * (1.f / E) - mean * mean;
    float rstd = rsqrtf(var + 1e-5f);
    #pragma unroll
    for (int i = 0; i < 4; ++i) {
        int idx = tid + i * 256;
        out[(size_t)row * E + idx] = (vals[i] - mean) * rstd * g[idx] + be[idx];
    }
}

// ---------- Poisson encode: spikes = (u < x) as bf16 ----------
__global__ __launch_bounds__(256)
void poisson_kernel(const floatx4* __restrict__ u, const floatx4* __restrict__ x,
                    bf16x4* __restrict__ sp)
{
    const int n4 = (2048 * 1024) / 4;
    int idx = blockIdx.x * 256 + threadIdx.x;
    int t = blockIdx.y;
    floatx4 uu = u[(size_t)t * n4 + idx];
    floatx4 xx = x[idx];
    bf16x4 o;
    #pragma unroll
    for (int i = 0; i < 4; ++i) o[i] = f2bf(uu[i] < xx[i] ? 1.f : 0.f);
    sp[(size_t)t * n4 + idx] = o;
}

// ---------- fused spike-GEMM + LIF scan + time-mean ----------
// WG tile 128(M) x 64(N); 8 waves (4 M x 2 N), wave tile 32x32.
// t loop outer (LIF state in VGPRs), K loop inner (LDS-staged WMMA via TDM).
__global__ __launch_bounds__(256)
void lif_spike_gemm_kernel(const bf16* __restrict__ spikes, const bf16* __restrict__ w1,
                           const float* __restrict__ b1, bf16* __restrict__ zmean)
{
    const int M = 2048, N = 4096, K = 1024, T = 16;
    __shared__ __attribute__((aligned(16))) bf16 As[128 * 32];
    __shared__ __attribute__((aligned(16))) bf16 Bs[64 * 32];
    const int tid = threadIdx.x, wave = tid >> 5, lane = tid & 31;
    const int wm = wave & 3, wn = wave >> 2;
    const int m0 = blockIdx.y * 128, n0 = blockIdx.x * 64;
    const int hf = lane >> 4, l = lane & 15;

    float vmem[2][2][8], icur[2][2][8], zsum[2][2][8];
    #pragma unroll
    for (int i = 0; i < 2; ++i)
        #pragma unroll
        for (int j = 0; j < 2; ++j)
            #pragma unroll
            for (int rr = 0; rr < 8; ++rr) { vmem[i][j][rr] = 0.f; icur[i][j][rr] = 0.f; zsum[i][j][rr] = 0.f; }

    float bj[2];
    bj[0] = b1[n0 + wn * 32 + l];
    bj[1] = b1[n0 + wn * 32 + 16 + l];

    const int ra = tid >> 1, ca = (tid & 1) * 16;
    const int rb = tid >> 2, cb = (tid & 3) * 8;

    for (int t = 0; t < T; ++t) {
        v8f acc[2][2];
        #pragma unroll
        for (int i = 0; i < 2; ++i) { acc[i][0] = v8zero(); acc[i][1] = v8zero(); }
        const bf16* sp = spikes + (size_t)t * M * K;

        for (int k0 = 0; k0 < K; k0 += 32) {
#if USE_TDM
            if (wave == 0) {
                tdm_load_2d_bf16(sp + (size_t)m0 * K + k0, lds_offset_of(As),
                                 128, 32, M - m0, K - k0, K);
                tdm_load_2d_bf16(w1 + (size_t)n0 * K + k0, lds_offset_of(Bs),
                                 64, 32, N - n0, K - k0, K);
                __builtin_amdgcn_s_wait_tensorcnt(0);
            }
            __syncthreads();
#else
            const bf16* ga = sp + (size_t)(m0 + ra) * K + k0 + ca;
            *(bf16x8*)(As + ra * 32 + ca)     = *(const bf16x8*)(ga);
            *(bf16x8*)(As + ra * 32 + ca + 8) = *(const bf16x8*)(ga + 8);
            *(bf16x8*)(Bs + rb * 32 + cb)     = *(const bf16x8*)(w1 + (size_t)(n0 + rb) * K + k0 + cb);
            __syncthreads();
#endif
            v16bf bfr[2];
            #pragma unroll
            for (int j = 0; j < 2; ++j)
                bfr[j] = load_bfrag(Bs + (wn * 32 + j * 16) * 32, 32, lane);
            #pragma unroll
            for (int i = 0; i < 2; ++i) {
                v16bf af = load_afrag(As + (wm * 32 + i * 16) * 32, 32, lane);
                #pragma unroll
                for (int j = 0; j < 2; ++j)
                    acc[i][j] = wmma_bf16(af, bfr[j], acc[i][j]);
            }
            __syncthreads();
        }

        // LIF step (norse lif_feed_forward_step), DT*TAU_MEM_INV = 0.1, 1-DT*TAU_SYN_INV = 0.8
        #pragma unroll
        for (int i = 0; i < 2; ++i)
            #pragma unroll
            for (int j = 0; j < 2; ++j)
                #pragma unroll
                for (int rr = 0; rr < 8; ++rr) {
                    float x_t = acc[i][j][rr] + bj[j];
                    float vd = vmem[i][j][rr] + 0.1f * (icur[i][j][rr] - vmem[i][j][rr]);
                    float id = icur[i][j][rr] * 0.8f;
                    float z = (vd - 1.0f) > 0.f ? 1.f : 0.f;
                    vmem[i][j][rr] = (1.f - z) * vd;
                    icur[i][j][rr] = id + x_t;
                    zsum[i][j][rr] += z;
                }
    }

    #pragma unroll
    for (int i = 0; i < 2; ++i)
        #pragma unroll
        for (int j = 0; j < 2; ++j)
            #pragma unroll
            for (int rr = 0; rr < 8; ++rr) {
                int mrow = m0 + wm * 32 + i * 16 + rr + hf * 8;
                int ncol = n0 + wn * 32 + j * 16 + l;
                zmean[(size_t)mrow * N + ncol] = f2bf(zsum[i][j][rr] * 0.0625f);
            }
}

// ---------- host orchestration ----------
extern "C" void kernel_launch(void* const* d_in, const int* in_sizes, int n_in,
                              void* d_out, int out_size, void* d_ws, size_t ws_size,
                              hipStream_t stream)
{
    (void)in_sizes; (void)n_in; (void)out_size; (void)ws_size;
    const float* src    = (const float*)d_in[0];
    const float* pois   = (const float*)d_in[1];
    const float* w_in   = (const float*)d_in[2];
    const float* b_in   = (const float*)d_in[3];
    const float* w_out  = (const float*)d_in[4];
    const float* b_out  = (const float*)d_in[5];
    const float* w1     = (const float*)d_in[6];
    const float* b1     = (const float*)d_in[7];
    const float* w2     = (const float*)d_in[8];
    const float* b2     = (const float*)d_in[9];
    const float* ln1_g  = (const float*)d_in[10];
    const float* ln1_b  = (const float*)d_in[11];
    const float* ln2_g  = (const float*)d_in[12];
    const float* ln2_b  = (const float*)d_in[13];
    float* out = (float*)d_out;

    const size_t Mr = 2048, E = 1024, E3 = 3072, FF = 4096, T = 16;

    char* ws = (char*)d_ws;
    size_t off = 0;
    auto take = [&](size_t bytes) -> char* {
        char* p = ws + off;
        off += (bytes + 255) & ~(size_t)255;
        return p;
    };
    bf16*  src_bf  = (bf16*) take(Mr * E  * 2);
    bf16*  win_bf  = (bf16*) take(E3 * E  * 2);
    bf16*  wout_bf = (bf16*) take(E  * E  * 2);
    bf16*  w1_bf   = (bf16*) take(FF * E  * 2);
    bf16*  w2_bf   = (bf16*) take(E  * FF * 2);
    bf16*  qkv_bf  = (bf16*) take(Mr * E3 * 2);
    bf16*  attn_bf = (bf16*) take(Mr * E  * 2);
    float* attnp   = (float*)take(Mr * E  * 4);
    float* xbuf    = (float*)take(Mr * E  * 4);
    bf16*  spikes  = (bf16*) take(T * Mr * E * 2);
    bf16*  zmean   = (bf16*) take(Mr * FF * 2);
    float* src2    = (float*)take(Mr * E  * 4);

    auto cvt = [&](const float* in, bf16* o, size_t n) {
        int n4 = (int)(n / 4);
        cvt_bf16_kernel<<<dim3((n4 + 255) / 256), 256, 0, stream>>>(
            (const floatx4*)in, (bf16x4*)o, n4);
    };
    // bf16 operand preparation
    cvt(src,   src_bf,  Mr * E);
    cvt(w_in,  win_bf,  E3 * E);
    cvt(w_out, wout_bf, E  * E);
    cvt(w1,    w1_bf,   FF * E);
    cvt(w2,    w2_bf,   E  * FF);

    // 1) qkv = src @ w_in^T + b_in  (output bf16 for attention)
    gemm_bf16_kernel<true><<<dim3(E3 / 128, Mr / 128), 256, 0, stream>>>(
        src_bf, win_bf, b_in, qkv_bf, (int)Mr, (int)E3, (int)E);

    // 2) fused per-head attention (scores, softmax, P@V)
    attn_kernel<<<dim3(8 * 16 * 4), 128, 0, stream>>>(qkv_bf, attn_bf);

    // 3) out projection
    gemm_bf16_kernel<false><<<dim3(E / 128, Mr / 128), 256, 0, stream>>>(
        attn_bf, wout_bf, b_out, attnp, (int)Mr, (int)E, (int)E);

    // 4) x = LN(src + attn_proj)
    ln_residual_kernel<<<dim3((unsigned)Mr), 256, 0, stream>>>(src, attnp, ln1_g, ln1_b, xbuf);

    // 5) Poisson encode -> bf16 spike tensor (L2-resident, 64 MB)
    poisson_kernel<<<dim3(2048, 16), 256, 0, stream>>>(
        (const floatx4*)pois, (const floatx4*)xbuf, (bf16x4*)spikes);

    // 6) fused spikes@w1^T + LIF scan + time-mean (dominant compute)
    lif_spike_gemm_kernel<<<dim3(FF / 64, Mr / 128), 256, 0, stream>>>(
        spikes, w1_bf, b1, zmean);

    // 7) src2 = zmean @ w2^T + b2
    gemm_bf16_kernel<false><<<dim3(E / 128, Mr / 128), 256, 0, stream>>>(
        zmean, w2_bf, b2, src2, (int)Mr, (int)E, (int)FF);

    // 8) out = LN(x + src2)
    ln_residual_kernel<<<dim3((unsigned)Mr), 256, 0, stream>>>(xbuf, src2, ln2_g, ln2_b, out);
}
